// EnhancedGNNModel_42709154791576
// MI455X (gfx1250) — compile-verified, matrix operand order checked by reading, not run on hardware
//
#include <hip/hip_runtime.h>
#include <hip/hip_bf16.h>

typedef __attribute__((ext_vector_type(16))) __bf16 v16bf;
typedef __attribute__((ext_vector_type(8)))  float  v8f;

#define GNN_N 50000
#define GNN_E 800000
#define GNN_H 256

// Toggle CDNA5 async global->LDS staging (ASYNCcnt path). Fallback = VGPR copy.
#define USE_ASYNC_LDS 1

// ---------------- elementwise / aggregation kernels ----------------

__global__ void count_edges_kernel(const int* __restrict__ dst, float* __restrict__ cnt, int E) {
    int e = blockIdx.x * blockDim.x + threadIdx.x;
    if (e < E) atomicAdd(&cnt[dst[e]], 1.0f);
}

__global__ void recip_cnt_kernel(float* __restrict__ cnt, int n) {
    int i = blockIdx.x * blockDim.x + threadIdx.x;
    if (i < n) cnt[i] = 1.0f / fmaxf(cnt[i], 1.0f);
}

// one thread per (edge, 4-float chunk): 64 chunks per edge (H=256)
__global__ void scatter_acc_kernel(const float* __restrict__ x, const int* __restrict__ src,
                                   const int* __restrict__ dst, float* __restrict__ agg, int E) {
    int gid = blockIdx.x * blockDim.x + threadIdx.x;
    int e = gid >> 6;
    if (e >= E) return;
    int c = (gid & 63) << 2;
    const float4 v = *(const float4*)(x + (size_t)src[e] * GNN_H + c);
    float* out = agg + (size_t)dst[e] * GNN_H + c;
    atomicAdd(out + 0, v.x);
    atomicAdd(out + 1, v.y);
    atomicAdd(out + 2, v.z);
    atomicAdd(out + 3, v.w);
}

// mean = agg * cnt_inv[row], converted to bf16 (feeds WMMA A-operand)
__global__ void mean_to_bf16_kernel(const float* __restrict__ agg, const float* __restrict__ cinv,
                                    __bf16* __restrict__ out, int total) {
    int i = blockIdx.x * blockDim.x + threadIdx.x;
    if (i < total) out[i] = (__bf16)(agg[i] * cinv[i >> 8]);   // H==256
}

__global__ void f32_to_bf16_kernel(const float* __restrict__ in, __bf16* __restrict__ out, int n) {
    int i = blockIdx.x * blockDim.x + threadIdx.x;
    if (i < n) out[i] = (__bf16)in[i];
}

// ---------------- BN (eval-style over batch) + ReLU ----------------

// blockDim.x == 256 (one column per thread), 128 rows per block
__global__ void bn_stats_kernel(const float* __restrict__ C, float* __restrict__ sum,
                                float* __restrict__ sumsq, int rows) {
    int col = threadIdx.x;
    int r0 = blockIdx.x * 128;
    int r1 = min(r0 + 128, rows);
    float s = 0.f, s2 = 0.f;
    for (int r = r0; r < r1; ++r) {
        float v = C[(size_t)r * GNN_H + col];
        s += v; s2 += v * v;
    }
    atomicAdd(&sum[col], s);
    atomicAdd(&sumsq[col], s2);
}

__global__ void bn_relu_kernel(float* __restrict__ C, const float* __restrict__ sum,
                               const float* __restrict__ sumsq, const float* __restrict__ g,
                               const float* __restrict__ bt, int rows) {
    int i = blockIdx.x * blockDim.x + threadIdx.x;
    int total = rows * GNN_H;
    if (i >= total) return;
    int col = i & (GNN_H - 1);
    float inv_n = 1.0f / (float)rows;
    float mu = sum[col] * inv_n;
    float var = sumsq[col] * inv_n - mu * mu;
    float v = (C[i] - mu) * rsqrtf(var + 1e-5f) * g[col] + bt[col];
    C[i] = fmaxf(v, 0.0f);
}

// ---------------- dual-GEMM via bf16 WMMA, LDS-staged A ----------------
// C[M x O] = A1[M x K] @ W1[O x K]^T + A2[M x K] @ W2[O x K]^T + bias[O]
// block = 128 threads (4 waves); each block: 16 rows x 64 cols of C.
// A1/A2 row-tile (16 x 256 each, 16 KB total) staged once in LDS via
// global_load_async_to_lds_b128 and shared by all 4 waves.
__global__ __launch_bounds__(128)
void gemm_dual_wmma_kernel(const __bf16* __restrict__ A1, const __bf16* __restrict__ A2,
                           const __bf16* __restrict__ W1, const __bf16* __restrict__ W2,
                           const float* __restrict__ bias, float* __restrict__ C,
                           int O, int ldC) {
    constexpr int K = GNN_H;                    // 256
    __shared__ __bf16 sA[2 * 16 * K];           // [A1 tile | A2 tile]

    const int m0  = blockIdx.x * 16;
    const int tid = threadIdx.x;

    // ---- cooperative stage: 1024 chunks of 16 B (512 per matrix) ----
    for (int c = tid; c < 1024; c += 128) {
        int mat = c >> 9;                       // 0 -> A1, 1 -> A2
        int cc  = c & 511;
        int row = cc >> 5;                      // 0..15
        int chk = cc & 31;                      // 16B chunk within 512B row
        const __bf16* g = (mat ? A2 : A1) + (size_t)(m0 + row) * K + chk * 8;
        __bf16* l = &sA[(mat * 16 + row) * K + chk * 8];
#if USE_ASYNC_LDS
        unsigned lds_off = (unsigned)(size_t)l;   // generic->LDS: low 32 bits
        asm volatile("global_load_async_to_lds_b128 %0, %1, off"
                     :: "v"(lds_off), "v"(g) : "memory");
#else
        *(float4*)l = *(const float4*)g;
#endif
    }
#if USE_ASYNC_LDS
    asm volatile("s_wait_asynccnt 0x0" ::: "memory");
#endif
    __syncthreads();

    const int wave = tid >> 5;
    const int lane = tid & 31;
    const int l15  = lane & 15;
    const int half = (lane >> 4) & 1;           // 0: lanes 0-15, 1: lanes 16-31
    const int kb   = half * 8;                  // per ISA 16-bit A/B layout

    const int n0    = blockIdx.y * 64 + wave * 16;
    const int ocol  = n0 + l15;
    const int osafe = (ocol < O) ? ocol : 0;    // clamp: junk cols never stored

    const __bf16* sA1 = &sA[l15 * K];
    const __bf16* sA2 = &sA[(16 + l15) * K];

    v8f c = {};
    for (int k0 = 0; k0 < K; k0 += 32) {
        const __bf16* a1p = sA1 + k0 + kb;
        const __bf16* a2p = sA2 + k0 + kb;
        const __bf16* w1p = W1 + (size_t)osafe * K + k0 + kb;
        const __bf16* w2p = W2 + (size_t)osafe * K + k0 + kb;
        v16bf a1, a2, b1, b2;
#pragma unroll
        for (int e = 0; e < 8; ++e) {
            a1[e] = a1p[e];  a1[8 + e] = a1p[16 + e];
            a2[e] = a2p[e];  a2[8 + e] = a2p[16 + e];
            b1[e] = w1p[e];  b1[8 + e] = w1p[16 + e];
            b2[e] = w2p[e];  b2[8 + e] = w2p[16 + e];
        }
        c = __builtin_amdgcn_wmma_f32_16x16x32_bf16(false, a1, false, b1, (short)0, c, false, false);
        c = __builtin_amdgcn_wmma_f32_16x16x32_bf16(false, a2, false, b2, (short)0, c, false, false);
    }

    if (ocol < O) {
        float bv = bias[ocol];
#pragma unroll
        for (int r = 0; r < 8; ++r) {
            int row = m0 + r + half * 8;
            C[(size_t)row * ldC + ocol] = c[r] + bv;
        }
    }
}

// ---------------- host orchestration ----------------

static void aggregate_stage(hipStream_t s, const float* h_in, const int* esrc, const int* edst,
                            float* agg, const float* cinv, __bf16* meanbf, __bf16* xbf) {
    hipMemsetAsync(agg, 0, (size_t)GNN_N * GNN_H * sizeof(float), s);
    scatter_acc_kernel<<<(GNN_E * 64) / 256, 256, 0, s>>>(h_in, esrc, edst, agg, GNN_E);
    mean_to_bf16_kernel<<<(GNN_N * GNN_H) / 256, 256, 0, s>>>(agg, cinv, meanbf, GNN_N * GNN_H);
    f32_to_bf16_kernel<<<(GNN_N * GNN_H) / 256, 256, 0, s>>>(h_in, xbf, GNN_N * GNN_H);
}

static void gemm_stage(hipStream_t s, const __bf16* meanbf, const __bf16* xbf,
                       const __bf16* wl, const __bf16* wr, const float* bias,
                       float* out, int O) {
    dim3 grid(GNN_N / 16, (O + 63) / 64);
    gemm_dual_wmma_kernel<<<grid, 128, 0, s>>>(meanbf, xbf, wl, wr, bias, out, O, O);
}

static void bn_stage(hipStream_t s, float* buf, float* stats, const float* g, const float* bt) {
    hipMemsetAsync(stats, 0, 2 * GNN_H * sizeof(float), s);
    bn_stats_kernel<<<(GNN_N + 127) / 128, 256, 0, s>>>(buf, stats, stats + GNN_H, GNN_N);
    bn_relu_kernel<<<(GNN_N * GNN_H + 255) / 256, 256, 0, s>>>(buf, stats, stats + GNN_H, g, bt, GNN_N);
}

extern "C" void kernel_launch(void* const* d_in, const int* in_sizes, int n_in,
                              void* d_out, int out_size, void* d_ws, size_t ws_size,
                              hipStream_t stream) {
    const float* x    = (const float*)d_in[0];
    const int*   esrc = (const int*)d_in[1];
    const int*   edst = (const int*)d_in[2];
    const float* w1l = (const float*)d_in[3];  const float* b1  = (const float*)d_in[4];
    const float* w1r = (const float*)d_in[5];  const float* g1  = (const float*)d_in[6];
    const float* bt1 = (const float*)d_in[7];
    const float* w2l = (const float*)d_in[8];  const float* b2  = (const float*)d_in[9];
    const float* w2r = (const float*)d_in[10]; const float* g2  = (const float*)d_in[11];
    const float* bt2 = (const float*)d_in[12];
    const float* w3l = (const float*)d_in[13]; const float* b3  = (const float*)d_in[14];
    const float* w3r = (const float*)d_in[15]; const float* g3  = (const float*)d_in[16];
    const float* bt3 = (const float*)d_in[17];
    const float* wal = (const float*)d_in[18]; const float* ba  = (const float*)d_in[19];
    const float* war = (const float*)d_in[20];
    const float* wsl = (const float*)d_in[21]; const float* bs  = (const float*)d_in[22];
    const float* wsr = (const float*)d_in[23];
    const float* wel = (const float*)d_in[24]; const float* be  = (const float*)d_in[25];
    const float* wer = (const float*)d_in[26];

    // workspace carve-up
    char* w = (char*)d_ws;
    float* agg = (float*)w;     w += (size_t)GNN_N * GNN_H * sizeof(float);
    float* hbuf = (float*)w;    w += (size_t)GNN_N * GNN_H * sizeof(float);
    __bf16* meanbf = (__bf16*)w; w += (size_t)GNN_N * GNN_H * sizeof(__bf16);
    __bf16* xbf = (__bf16*)w;    w += (size_t)GNN_N * GNN_H * sizeof(__bf16);
    float* cinv = (float*)w;    w += (size_t)GNN_N * sizeof(float);
    float* stats = (float*)w;   w += 2 * GNN_H * sizeof(float);
    __bf16* bw1l = (__bf16*)w;  w += (size_t)GNN_H * GNN_H * sizeof(__bf16);
    __bf16* bw1r = (__bf16*)w;  w += (size_t)GNN_H * GNN_H * sizeof(__bf16);
    __bf16* bw2l = (__bf16*)w;  w += (size_t)GNN_H * GNN_H * sizeof(__bf16);
    __bf16* bw2r = (__bf16*)w;  w += (size_t)GNN_H * GNN_H * sizeof(__bf16);
    __bf16* bw3l = (__bf16*)w;  w += (size_t)GNN_H * GNN_H * sizeof(__bf16);
    __bf16* bw3r = (__bf16*)w;  w += (size_t)GNN_H * GNN_H * sizeof(__bf16);
    __bf16* bwal = (__bf16*)w;  w += (size_t)21 * GNN_H * sizeof(__bf16);
    __bf16* bwar = (__bf16*)w;  w += (size_t)21 * GNN_H * sizeof(__bf16);
    __bf16* bwsl = (__bf16*)w;  w += (size_t)2 * GNN_H * sizeof(__bf16);
    __bf16* bwsr = (__bf16*)w;  w += (size_t)2 * GNN_H * sizeof(__bf16);
    __bf16* bwel = (__bf16*)w;  w += (size_t)5 * GNN_H * sizeof(__bf16);
    __bf16* bwer = (__bf16*)w;  w += (size_t)5 * GNN_H * sizeof(__bf16);

    // degree -> reciprocal counts (computed once per call)
    hipMemsetAsync(cinv, 0, (size_t)GNN_N * sizeof(float), stream);
    count_edges_kernel<<<(GNN_E + 255) / 256, 256, 0, stream>>>(edst, cinv, GNN_E);
    recip_cnt_kernel<<<(GNN_N + 255) / 256, 256, 0, stream>>>(cinv, GNN_N);

    // weight conversions to bf16
    #define CVT(src, dst, n) f32_to_bf16_kernel<<<((n) + 255) / 256, 256, 0, stream>>>(src, dst, n)
    CVT(w1l, bw1l, GNN_H * GNN_H); CVT(w1r, bw1r, GNN_H * GNN_H);
    CVT(w2l, bw2l, GNN_H * GNN_H); CVT(w2r, bw2r, GNN_H * GNN_H);
    CVT(w3l, bw3l, GNN_H * GNN_H); CVT(w3r, bw3r, GNN_H * GNN_H);
    CVT(wal, bwal, 21 * GNN_H);    CVT(war, bwar, 21 * GNN_H);
    CVT(wsl, bwsl, 2 * GNN_H);     CVT(wsr, bwsr, 2 * GNN_H);
    CVT(wel, bwel, 5 * GNN_H);     CVT(wer, bwer, 5 * GNN_H);
    #undef CVT

    // layer 1: x -> hbuf
    aggregate_stage(stream, x, esrc, edst, agg, cinv, meanbf, xbf);
    gemm_stage(stream, meanbf, xbf, bw1l, bw1r, b1, hbuf, GNN_H);
    bn_stage(stream, hbuf, stats, g1, bt1);

    // layer 2: hbuf -> hbuf (GEMM reads only bf16 staging; safe in-place)
    aggregate_stage(stream, hbuf, esrc, edst, agg, cinv, meanbf, xbf);
    gemm_stage(stream, meanbf, xbf, bw2l, bw2r, b2, hbuf, GNN_H);
    bn_stage(stream, hbuf, stats, g2, bt2);

    // layer 3
    aggregate_stage(stream, hbuf, esrc, edst, agg, cinv, meanbf, xbf);
    gemm_stage(stream, meanbf, xbf, bw3l, bw3r, b3, hbuf, GNN_H);
    bn_stage(stream, hbuf, stats, g3, bt3);

    // heads share one aggregation of h3
    aggregate_stage(stream, hbuf, esrc, edst, agg, cinv, meanbf, xbf);
    float* out = (float*)d_out;
    gemm_stage(stream, meanbf, xbf, bwal, bwar, ba, out, 21);                       // age [N,21]
    gemm_stage(stream, meanbf, xbf, bwsl, bwsr, bs, out + (size_t)GNN_N * 21, 2);   // sex [N,2]
    gemm_stage(stream, meanbf, xbf, bwel, bwer, be, out + (size_t)GNN_N * 23, 5);   // eth [N,5]
}
